// KohonenMap_60266981097662
// MI455X (gfx1250) — matrix operand order repaired and checked by reading, compiled, not run on hardware
//
#include <hip/hip_runtime.h>
#include <hip/hip_bf16.h>

// ---------------------------------------------------------------------------
// Kohonen-map nearest-node search for gfx1250 (MI455X), fused GEMM+argmin:
//   argmin_k |x_b - n_k|  ==  argmax-free  argmin_k ( |n_k|^2 - 2 <x_b,n_k> )
// Pass 1: nodes -> bf16 hi/lo split + |n|^2 into workspace (once).
// Pass 2: x @ nodes^T on v_wmma_f32_16x16x32_bf16 (3-term hi/lo product for
//         ~fp32-faithful argmin), nodes staged into LDS with
//         global_load_async_to_lds_b128 (ASYNCcnt), fused running argmin,
//         cooperative gather of winning node vectors.
// ---------------------------------------------------------------------------

typedef __attribute__((ext_vector_type(16))) __bf16 v16bf;
typedef __attribute__((ext_vector_type(8)))  float  v8f;
typedef __attribute__((ext_vector_type(4)))  float  v4f;

#define B_ROWS          65536
#define KNODES          4096
#define DIM             64
#define ROWS_PER_BLOCK  128       // 8 waves x 16 rows
#define NC              128       // nodes staged per LDS chunk
#define NCHUNK          (KNODES / NC)
#define NTHREADS        256

// ---- Pass 1: fp32 nodes -> bf16 hi/lo + per-node |n|^2 --------------------
__global__ __launch_bounds__(256)
void prep_nodes_kernel(const float* __restrict__ nodes,
                       __bf16* __restrict__ wsHi,
                       __bf16* __restrict__ wsLo,
                       float*  __restrict__ wsN2)
{
    const int t  = blockIdx.x * 256 + threadIdx.x;  // 16384 threads total
    const int e0 = t * 16;                          // 16 elements per thread
    float part = 0.0f;
    #pragma unroll
    for (int j = 0; j < 4; ++j) {
        const v4f f4 = *(const v4f*)(nodes + e0 + j * 4);
        #pragma unroll
        for (int u = 0; u < 4; ++u) {
            const float f  = f4[u];
            const __bf16 h = (__bf16)f;
            wsHi[e0 + j * 4 + u] = h;
            wsLo[e0 + j * 4 + u] = (__bf16)(f - (float)h);
            part = fmaf(f, f, part);
        }
    }
    // 4 consecutive lanes cover one node's 64 elements
    part += __shfl_xor(part, 1, 32);
    part += __shfl_xor(part, 2, 32);
    if ((t & 3) == 0) wsN2[t >> 2] = part;
}

// ---- Pass 2: fused GEMM + argmin + gather ---------------------------------
__global__ __launch_bounds__(NTHREADS)
void kohonen_argmin_kernel(const float*  __restrict__ x,
                           const float*  __restrict__ nodes,
                           const __bf16* __restrict__ wsHi,
                           const __bf16* __restrict__ wsLo,
                           const float*  __restrict__ wsN2,
                           float* __restrict__ outRepr,
                           float* __restrict__ outIdx)
{
    __shared__ __align__(32) __bf16 sHi[NC * DIM];   // 16 KB
    __shared__ __align__(32) __bf16 sLo[NC * DIM];   // 16 KB
    __shared__ __align__(16) float  sN2[NC];         // 512 B
    __shared__ int sWin[ROWS_PER_BLOCK];             // 512 B

    const int tid    = threadIdx.x;
    const int lane   = tid & 31;
    const int wave   = tid >> 5;
    const int l16    = lane & 15;
    const int hiHalf = lane >> 4;                    // 0: lanes 0-15, 1: 16-31

    const int rowBase = blockIdx.x * ROWS_PER_BLOCK;
    const int waveRow = rowBase + wave * 16;

    // ---- A fragments: this wave's 16 x-rows, bf16 hi/lo, two K-steps ------
    // A k order used here (contiguous, same as B): lanes 0-15 hold M=lane, K in {kb..kb+15}
    // with kb = kstep*32 (+16 for lanes 16-31).
    v16bf aHi[2], aLo[2];
    {
        const float* xr = x + (size_t)(waveRow + l16) * DIM;
        #pragma unroll
        for (int kk = 0; kk < 2; ++kk) {
            const int kb = kk * 32 + hiHalf * 16;
            #pragma unroll
            for (int i = 0; i < 16; ++i) {
                const int k = kb + i;
                const float f  = xr[k];
                const __bf16 h = (__bf16)f;
                aHi[kk][i] = h;
                aLo[kk][i] = (__bf16)(f - (float)h);
            }
        }
    }

    // ---- running per-lane (min score, argmin) for the 8 C-rows ------------
    float bestS[8];
    int   bestI[8];
    #pragma unroll
    for (int r = 0; r < 8; ++r) { bestS[r] = 3.0e38f; bestI[r] = 0; }

    // async-copy source bases (GVS mode: 64-bit SGPR base + 32-bit VGPR off)
    const __bf16* hiB = wsHi;
    const __bf16* loB = wsLo;
    const float*  n2B = wsN2;

    for (int c = 0; c < NCHUNK; ++c) {
        __syncthreads();   // previous chunk fully consumed before restage

        // ---- stage chunk: pure async byte-copy global -> LDS --------------
        // 16 KB hi + 16 KB lo: 64 B per thread each (4x B128, offset applies
        // to BOTH lds and global address per ISA 08_async_tensor 4.4).
        {
            const unsigned ldsHi = (unsigned)(uintptr_t)(sHi + tid * 32);
            const unsigned ldsLo = (unsigned)(uintptr_t)(sLo + tid * 32);
            const unsigned voff  = (unsigned)(c * (NC * DIM * 2) + tid * 64);
            asm volatile(
                "global_load_async_to_lds_b128 %0, %1, %2 offset:0\n\t"
                "global_load_async_to_lds_b128 %0, %1, %2 offset:16\n\t"
                "global_load_async_to_lds_b128 %0, %1, %2 offset:32\n\t"
                "global_load_async_to_lds_b128 %0, %1, %2 offset:48"
                :: "v"(ldsHi), "v"(voff), "s"(hiB) : "memory");
            asm volatile(
                "global_load_async_to_lds_b128 %0, %1, %2 offset:0\n\t"
                "global_load_async_to_lds_b128 %0, %1, %2 offset:16\n\t"
                "global_load_async_to_lds_b128 %0, %1, %2 offset:32\n\t"
                "global_load_async_to_lds_b128 %0, %1, %2 offset:48"
                :: "v"(ldsLo), "v"(voff), "s"(loB) : "memory");
            if (wave == 0) {   // wave-uniform branch: 512 B of |n|^2
                const unsigned ldsN2 = (unsigned)(uintptr_t)(sN2 + lane * 4);
                const unsigned voffN = (unsigned)(c * (NC * 4) + lane * 16);
                asm volatile(
                    "global_load_async_to_lds_b128 %0, %1, %2 offset:0"
                    :: "v"(ldsN2), "v"(voffN), "s"(n2B) : "memory");
            }
        }

        if (c + 1 < NCHUNK)   // uniform; warm GL2 for the next chunk
            __builtin_prefetch(hiB + (size_t)(c + 1) * NC * DIM + tid * 32, 0, 2);

        asm volatile("s_wait_asynccnt 0x0" ::: "memory");
        __syncthreads();

        // ---- 8 column tiles of 16 nodes, 6 WMMAs each ---------------------
        #pragma unroll 1
        for (int ct = 0; ct < NC / 16; ++ct) {
            // B layout: lane holds node (ct*16 + l16), contiguous K range
            // [hiHalf*16 .. +16) per K-step -> one 32B LDS vector load.
            const __bf16* bh = sHi + (ct * 16 + l16) * DIM + hiHalf * 16;
            const __bf16* bl = sLo + (ct * 16 + l16) * DIM + hiHalf * 16;
            const v16bf b0h = *(const v16bf*)(bh);
            const v16bf b1h = *(const v16bf*)(bh + 32);
            const v16bf b0l = *(const v16bf*)(bl);
            const v16bf b1l = *(const v16bf*)(bl + 32);

            v8f acc = {};
            acc = __builtin_amdgcn_wmma_f32_16x16x32_bf16(false, aHi[0], false, b0h, (short)0, acc, false, false);
            acc = __builtin_amdgcn_wmma_f32_16x16x32_bf16(false, aHi[1], false, b1h, (short)0, acc, false, false);
            acc = __builtin_amdgcn_wmma_f32_16x16x32_bf16(false, aHi[0], false, b0l, (short)0, acc, false, false);
            acc = __builtin_amdgcn_wmma_f32_16x16x32_bf16(false, aHi[1], false, b1l, (short)0, acc, false, false);
            acc = __builtin_amdgcn_wmma_f32_16x16x32_bf16(false, aLo[0], false, b0h, (short)0, acc, false, false);
            acc = __builtin_amdgcn_wmma_f32_16x16x32_bf16(false, aLo[1], false, b1h, (short)0, acc, false, false);

            const float n2v    = sN2[ct * 16 + l16];
            const int  nodeIdx = c * NC + ct * 16 + l16;
            #pragma unroll
            for (int r = 0; r < 8; ++r) {
                const float score = fmaf(-2.0f, acc[r], n2v);
                // strict '<' keeps the first (lowest) index within a lane
                if (score < bestS[r]) { bestS[r] = score; bestI[r] = nodeIdx; }
            }
        }
    }

    // ---- cross-lane argmin within each 16-lane half (N dimension) ---------
    // C layout: lanes 0-15 -> row waveRow+r ; lanes 16-31 -> row waveRow+8+r.
    #pragma unroll
    for (int r = 0; r < 8; ++r) {
        float s = bestS[r];
        int   i = bestI[r];
        #pragma unroll
        for (int off = 8; off >= 1; off >>= 1) {
            const float os = __shfl_xor(s, off, 16);
            const int   oi = __shfl_xor(i, off, 16);
            if (os < s || (os == s && oi < i)) { s = os; i = oi; }
        }
        if (l16 == 0) sWin[wave * 16 + hiHalf * 8 + r] = i;
    }
    __syncthreads();

    // ---- cooperative gather: node_repr + indices (coalesced float4s) ------
    {
        const int row  = tid >> 1;        // 0..127
        const int half = tid & 1;         // which 32-element half of DIM
        const int widx = sWin[row];
        const int grow = rowBase + row;
        const float* src = nodes + (size_t)widx * DIM + half * 32;
        float*       dst = outRepr + (size_t)grow * DIM + half * 32;
        #pragma unroll
        for (int j = 0; j < 8; ++j)
            *(v4f*)(dst + j * 4) = *(const v4f*)(src + j * 4);
        if (half == 0) outIdx[grow] = (float)widx;   // <= 4095, exact in f32
    }
}

extern "C" void kernel_launch(void* const* d_in, const int* in_sizes, int n_in,
                              void* d_out, int out_size, void* d_ws, size_t ws_size,
                              hipStream_t stream)
{
    (void)in_sizes; (void)n_in; (void)out_size; (void)ws_size;
    const float* x     = (const float*)d_in[0];
    const float* nodes = (const float*)d_in[1];

    // workspace layout: [512 KB bf16 hi][512 KB bf16 lo][16 KB f32 n^2]
    __bf16* wsHi = (__bf16*)d_ws;
    __bf16* wsLo = (__bf16*)((char*)d_ws + (size_t)KNODES * DIM * 2);
    float*  wsN2 = (float*) ((char*)d_ws + (size_t)KNODES * DIM * 4);

    float* outRepr = (float*)d_out;                        // [65536 x 64]
    float* outIdx  = (float*)d_out + (size_t)B_ROWS * DIM; // [65536]

    prep_nodes_kernel<<<KNODES * DIM / (256 * 16), 256, 0, stream>>>(
        nodes, wsHi, wsLo, wsN2);
    kohonen_argmin_kernel<<<B_ROWS / ROWS_PER_BLOCK, NTHREADS, 0, stream>>>(
        x, nodes, wsHi, wsLo, wsN2, outRepr, outIdx);
}
